// KNN_74577812127972
// MI455X (gfx1250) — matrix operand order repaired and checked by reading, not compiled, not run on hardware
//
#include <hip/hip_runtime.h>

typedef __attribute__((ext_vector_type(16))) _Float16 v16h;
typedef __attribute__((ext_vector_type(8)))  _Float16 v8h;
typedef __attribute__((ext_vector_type(4)))  _Float16 v4h;
typedef __attribute__((ext_vector_type(8)))  float    v8f;

#define KNN   9
#define DIMS  128
#define QT    2      // query tiles (of 16) per block -> 32 queries/block
#define WAVES 16     // waves per block (train-tile slices), blockDim = 512
#define BIGF  3.0e38f

// ---- pre-pass: fp32 -> f16 (optionally scaled) + row sq-norms, with padding --
// Rows in [nrows, nrowsPad) are zero-filled with norm=BIGF so padded train rows
// can never enter any top-9 list (removes all bounds checks from the hot loop).
__global__ __launch_bounds__(256)
void knn_convert(const float* __restrict__ src, _Float16* __restrict__ dst,
                 float* __restrict__ norms, float scale, int nrows, int nrowsPad) {
  const int wave = threadIdx.x >> 5;
  const int lane = threadIdx.x & 31;
  const int row  = blockIdx.x * 8 + wave;          // 8 waves / block, 1 row / wave
  if (row >= nrowsPad) return;                     // wave-uniform exit
  if (row >= nrows) {                              // padding row
    v4h z = {};
    *(v4h*)(dst + (size_t)row * DIMS + lane * 4) = z;
    if (norms != nullptr && lane == 0) norms[row] = BIGF;
    return;
  }
  const float4 f = *(const float4*)(src + (size_t)row * DIMS + lane * 4);
  float ss = f.x * f.x + f.y * f.y + f.z * f.z + f.w * f.w;
  v4h h;
  h.x = (_Float16)(scale * f.x);
  h.y = (_Float16)(scale * f.y);
  h.z = (_Float16)(scale * f.z);
  h.w = (_Float16)(scale * f.w);
  *(v4h*)(dst + (size_t)row * DIMS + lane * 4) = h;
  if (norms != nullptr) {
    #pragma unroll
    for (int off = 16; off > 0; off >>= 1) ss += __shfl_xor(ss, off, 32);
    if (lane == 0) norms[row] = ss;
  }
}

// ---------------- sorted top-9 insertion (registers, fully unrolled) ---------
__device__ __forceinline__ void ins9(float d, int j, float (&dd)[KNN], int (&ii)[KNN]) {
  if (d < dd[KNN - 1]) {
    dd[KNN - 1] = d;
    ii[KNN - 1] = j;
    #pragma unroll
    for (int i = KNN - 1; i > 0; --i) {
      if (dd[i] < dd[i - 1]) {
        float tf = dd[i - 1]; dd[i - 1] = dd[i]; dd[i] = tf;
        int   ti = ii[i - 1]; ii[i - 1] = ii[i]; ii[i] = ti;
      }
    }
  }
}

// ---------------- main fused kernel: WMMA distances + top-9 + vote -----------
__global__ __launch_bounds__(WAVES * 32)
void knn_main(const _Float16* __restrict__ Xtr,   // [nTrainPad][128] f16 (padded)
              const _Float16* __restrict__ Xte,   // [nTest][128] f16, scaled by -2
              const float*    __restrict__ tn,    // [nTrainPad] fp32 sq-norms
              const int*      __restrict__ ytr,   // [nTrain]
              int*            __restrict__ out,   // [nTest] predictions
              int nTrainPad, int nTest) {
  const int lane  = threadIdx.x & 31;
  const int wave  = threadIdx.x >> 5;
  const int hi    = lane >> 4;       // lane half-group
  const int ln    = lane & 15;
  const int qbase = blockIdx.x * (QT * 16);

  // ---- preload B fragments (queries), reused for the whole train sweep ----
  // B layout: lane holds column N=ln, K = hi*16 + kc*32 .. +15 (contiguous).
  v16h bq[QT][4];
  #pragma unroll
  for (int t = 0; t < QT; ++t) {
    int q = qbase + t * 16 + ln;
    if (q >= nTest) q = nTest - 1;
    const _Float16* qp = Xte + (size_t)q * DIMS + hi * 16;
    #pragma unroll
    for (int kc = 0; kc < 4; ++kc) bq[t][kc] = *(const v16h*)(qp + kc * 32);
  }

  float dist[QT][KNN];
  int   idx [QT][KNN];
  #pragma unroll
  for (int t = 0; t < QT; ++t)
    #pragma unroll
    for (int i = 0; i < KNN; ++i) { dist[t][i] = BIGF; idx[t][i] = -1; }

  const int ntiles = nTrainPad >> 4;
  for (int tile = wave; tile < ntiles; tile += WAVES) {
    const int tb = tile << 4;
    // ---- A fragments: lane holds train row tb+ln, K-base hi*16 ------------
    const _Float16* ap = Xtr + (size_t)(tb + ln) * DIMS + hi * 16;
    v16h a[4];
    #pragma unroll
    for (int kc = 0; kc < 4; ++kc) {
      v8h lo  = *(const v8h*)(ap + kc * 32);
      v8h hi8 = *(const v8h*)(ap + kc * 32 + 8);
      a[kc] = __builtin_shufflevector(lo, hi8, 0, 1, 2, 3, 4, 5, 6, 7,
                                      8, 9, 10, 11, 12, 13, 14, 15);
    }
    // WGP-scope prefetch of the next train tile (builtin only reaches SE/SYS
    // scope, which skips the WGP cache; scope field 0 == WGP is the useful
    // one for L2-resident data). Address always stays inside the workspace
    // (Xte16/tn regions follow Xtr16), so no guard needed.
    {
      const unsigned long long pa = (unsigned long long)
          (Xtr + (size_t)(((size_t)(tile + WAVES) << 4) + ln) * DIMS + hi * 64);
      asm volatile("global_prefetch_b8 %0, off" :: "v"(pa));
    }

    // fp32 sq-norms for this lane's 8 output rows (tb + hi*8 .. +7)
    const float4 t0 = *(const float4*)(tn + tb + hi * 8);
    const float4 t1 = *(const float4*)(tn + tb + hi * 8 + 4);
    const float tv[8] = {t0.x, t0.y, t0.z, t0.w, t1.x, t1.y, t1.z, t1.w};

    // ---- two independent WMMA accumulation chains, interleaved ------------
    v8f acc[QT];
    #pragma unroll
    for (int t = 0; t < QT; ++t) acc[t] = (v8f){};
    #pragma unroll
    for (int kc = 0; kc < 4; ++kc)
      #pragma unroll
      for (int t = 0; t < QT; ++t)
        acc[t] = __builtin_amdgcn_wmma_f32_16x16x32_f16(
            false, a[kc], false, bq[t][kc], (short)0, acc[t], false, false);

    // D layout: lane -> query ln, rows tb + hi*8 + r. score = tn[j] - 2 q.t
    #pragma unroll
    for (int t = 0; t < QT; ++t)
      #pragma unroll
      for (int r = 0; r < 8; ++r) {
        const int j = tb + hi * 8 + r;
        const float s = acc[t][r] + tv[r];
        ins9(s, j, dist[t], idx[t]);       // padded rows carry BIGF: never enter
      }
  }

  // ---- merge lane partner (same query, other train-row half) --------------
  #pragma unroll
  for (int t = 0; t < QT; ++t)
    #pragma unroll
    for (int i = 0; i < KNN; ++i) {
      const float pd = __shfl_xor(dist[t][i], 16, 32);
      const int   pj = __shfl_xor(idx[t][i], 16, 32);
      if (hi == 0) ins9(pd, pj, dist[t], idx[t]);
    }

  // ---- cross-wave merge via LDS -------------------------------------------
  __shared__ float sD[QT * 16][WAVES][KNN];
  __shared__ int   sI[QT * 16][WAVES][KNN];
  if (hi == 0) {
    #pragma unroll
    for (int t = 0; t < QT; ++t)
      #pragma unroll
      for (int i = 0; i < KNN; ++i) {
        sD[t * 16 + ln][wave][i] = dist[t][i];
        sI[t * 16 + ln][wave][i] = idx[t][i];
      }
  }
  __syncthreads();

  if (threadIdx.x < QT * 16) {
    const int ql = threadIdx.x;
    float bd[KNN]; int bi[KNN];
    #pragma unroll
    for (int i = 0; i < KNN; ++i) { bd[i] = BIGF; bi[i] = -1; }
    for (int w = 0; w < WAVES; ++w)
      #pragma unroll
      for (int i = 0; i < KNN; ++i) ins9(sD[ql][w][i], sI[ql][w][i], bd, bi);

    // 3-class majority vote, first-max tie-break (matches jnp.argmax)
    int c0 = 0, c1 = 0, c2 = 0;
    #pragma unroll
    for (int i = 0; i < KNN; ++i) {
      const int lab = (bi[i] >= 0) ? ytr[bi[i]] : 0;
      c0 += (lab == 0); c1 += (lab == 1); c2 += (lab == 2);
    }
    int pred = 0, best = c0;
    if (c1 > best) { best = c1; pred = 1; }
    if (c2 > best) { pred = 2; }
    const int q = qbase + ql;
    if (q < nTest) out[q] = pred;
  }
}

// ---------------- host-side launcher -----------------------------------------
extern "C" void kernel_launch(void* const* d_in, const int* in_sizes, int n_in,
                              void* d_out, int out_size, void* d_ws, size_t ws_size,
                              hipStream_t stream) {
  const float* Xtr = (const float*)d_in[0];
  const int*   ytr = (const int*)d_in[1];
  const float* Xte = (const float*)d_in[2];
  const int nTrain    = in_sizes[0] / DIMS;
  const int nTest     = in_sizes[2] / DIMS;
  const int nTrainPad = (nTrain + 15) & ~15;

  // workspace layout (all 256B-aligned for these sizes)
  char* ws = (char*)d_ws;
  _Float16* Xtr16 = (_Float16*)ws;
  _Float16* Xte16 = (_Float16*)(ws + (size_t)nTrainPad * DIMS * 2);
  float*    tn    = (float*)(ws + (size_t)nTrainPad * DIMS * 2 + (size_t)nTest * DIMS * 2);

  // pre-pass: f16 conversion (+ train norms); queries pre-scaled by -2
  knn_convert<<<(nTrainPad + 7) / 8, 256, 0, stream>>>(Xtr, Xtr16, tn, 1.0f,
                                                       nTrain, nTrainPad);
  knn_convert<<<(nTest + 7) / 8, 256, 0, stream>>>(Xte, Xte16, nullptr, -2.0f,
                                                   nTest, nTest);

  // fused WMMA distance + top-9 + vote
  const int blocks = (nTest + QT * 16 - 1) / (QT * 16);
  knn_main<<<blocks, WAVES * 32, 0, stream>>>(Xtr16, Xte16, tn, ytr,
                                              (int*)d_out, nTrainPad, nTest);
}
